// RNN_64879775973779
// MI455X (gfx1250) — matrix operand order repaired and checked by reading, not compiled, not run on hardware
//
#include <hip/hip_runtime.h>

// ---------------------------------------------------------------------------
// GRU (B=128, T=1024, IN=256, H=512) persistent-scan kernel for gfx1250.
//
//   * One-time parallel pass converts x to bf16 (off the critical path);
//     the sequential scan then does pure bf16 WMMA with no per-step cvt VALU.
//   * 16 persistent workgroups (256 thr = 8 wave32), each owns 32 h-columns
//     -> 96 gate rows of W_ih/W_hh pinned in LDS as bf16 (144 KB / 320 KB).
//   * Per timestep: fused GEMM [x_t | h] x [W_ih | W_hh]^T via
//     v_wmma_f32_16x16x32_bf16 (fp32 accum), fp32 gate math,
//     h exchanged through L2 (double buffered), one grid barrier per step.
// ---------------------------------------------------------------------------

typedef __attribute__((ext_vector_type(16))) __bf16 v16bf;
typedef __attribute__((ext_vector_type(8)))  float  v8f;
typedef unsigned short u16;

constexpr int kB   = 128;
constexpr int kT   = 1024;
constexpr int kIN  = 256;
constexpr int kH   = 512;
constexpr int kG   = 16;     // workgroups
constexpr int kHs  = 32;     // h-columns per WG
constexpr int kGC  = 96;     // gate rows per WG (3*kHs)

// fp32 -> bf16, round-to-nearest-even
__device__ __forceinline__ u16 f2bf(float f) {
  unsigned u = __float_as_uint(f);
  unsigned r = u + 0x7FFFu + ((u >> 16) & 1u);
  return (u16)(r >> 16);
}

union BF16x16 {         // one WMMA A/B fragment per lane (8 VGPRs)
  uint4 q[2];
  u16   h[16];
  v16bf v;
};

union BF16x8 {
  uint4 q;
  u16   h[8];
};

__device__ __forceinline__ float sigmoidf(float x) {
  return 1.0f / (1.0f + __expf(-x));
}

// gate row j in [0,96) of workgroup wg -> row of the (3H, K) weight matrix
__device__ __forceinline__ int gate_row(int wg, int j) {
  const int c0 = wg * kHs;
  return (j < 32) ? (c0 + j)
       : (j < 64) ? (kH + c0 + (j - 32))
                  : (2 * kH + c0 + (j - 64));
}

// ---------------------------------------------------------------------------
// Kernel 0: convert x (B,T,IN) fp32 -> bf16, 8 elements per thread.
// One-time, bandwidth-bound, fully parallel (~9 us at 23.3 TB/s).
// ---------------------------------------------------------------------------
__global__ void gru_x_to_bf16(const float* __restrict__ x,
                              u16* __restrict__ xb) {
  const size_t i = ((size_t)blockIdx.x * blockDim.x + threadIdx.x) * 8;
  const float4 f0 = *(const float4*)(x + i);
  const float4 f1 = *(const float4*)(x + i + 4);
  BF16x8 o;
  o.h[0] = f2bf(f0.x); o.h[1] = f2bf(f0.y); o.h[2] = f2bf(f0.z); o.h[3] = f2bf(f0.w);
  o.h[4] = f2bf(f1.x); o.h[5] = f2bf(f1.y); o.h[6] = f2bf(f1.z); o.h[7] = f2bf(f1.w);
  *(uint4*)(xb + i) = o.q;
}

// ---------------------------------------------------------------------------
// Kernel 1: repack W_ih / W_hh rows per-WG and convert fp32 -> bf16.
// ---------------------------------------------------------------------------
__global__ void gru_pack_weights(const float* __restrict__ W_ih,
                                 const float* __restrict__ W_hh,
                                 u16* __restrict__ wih_pk,   // [16][96][256]
                                 u16* __restrict__ whh_pk) { // [16][96][512]
  const int i = blockIdx.x * blockDim.x + threadIdx.x;
  constexpr int NHH = kG * kGC * kH;
  constexpr int NIH = kG * kGC * kIN;
  if (i < NHH) {
    const int wg = i / (kGC * kH);
    const int j  = (i / kH) % kGC;
    const int k  = i % kH;
    whh_pk[i] = f2bf(W_hh[(size_t)gate_row(wg, j) * kH + k]);
  }
  if (i < NIH) {
    const int wg = i / (kGC * kIN);
    const int j  = (i / kIN) % kGC;
    const int k  = i % kIN;
    wih_pk[i] = f2bf(W_ih[(size_t)gate_row(wg, j) * kIN + k]);
  }
}

// ---------------------------------------------------------------------------
// Kernel 2: persistent GRU scan.
// ---------------------------------------------------------------------------
__global__ __launch_bounds__(256, 1)
void gru_scan(const u16*  __restrict__ xb,        // (B,T,IN) bf16
              const float* __restrict__ bias,     // (3H)
              const float* __restrict__ bias_n,   // (H)
              const u16*  __restrict__ wih_pk,    // [16][96][256] bf16
              const u16*  __restrict__ whh_pk,    // [16][96][512] bf16
              float* __restrict__ hf,             // [2][B][H] fp32 master
              u16*   __restrict__ hb,             // [2][B][H] bf16 shadow
              unsigned* __restrict__ syncp) {
  extern __shared__ u16 lds[];
  u16* whh_l = lds;                  // [96][512] bf16 (96 KB)
  u16* wih_l = lds + kGC * kH;       // [96][256] bf16 (48 KB)

  const int wg   = blockIdx.x;       // 0..15
  const int c0   = wg * kHs;
  const int tid  = threadIdx.x;
  const int lane = tid & 31;
  const int wave = tid >> 5;         // 0..7 -> 16 batch rows each
  const int hi   = lane >> 4;        // half-wave select
  const int lm   = lane & 15;        // M (A-frag) / N (B-frag, D tile)
  const int mA   = wave * 16 + lm;   // batch row this lane sources for A

  // ---- stage this WG's weight slices into LDS (uint4 = 8 bf16) ----
  {
    const uint4* s0 = (const uint4*)(whh_pk + (size_t)wg * kGC * kH);
    uint4* d0 = (uint4*)whh_l;
    for (int i = tid; i < kGC * kH / 8; i += 256) d0[i] = s0[i];
    const uint4* s1 = (const uint4*)(wih_pk + (size_t)wg * kGC * kIN);
    uint4* d1 = (uint4*)wih_l;
    for (int i = tid; i < kGC * kIN / 8; i += 256) d1[i] = s1[i];
  }
  __syncthreads();

  // ---- per-lane bias registers (D-tile column = c0 + 16*tau + lm) ----
  float brg[2], bzg[2], bng[2], bnn[2];
#pragma unroll
  for (int tau = 0; tau < 2; ++tau) {
    const int col = c0 + 16 * tau + lm;
    brg[tau] = bias[col];
    bzg[tau] = bias[kH + col];
    bng[tau] = bias[2 * kH + col];
    bnn[tau] = bias_n[col];
  }
  const int mrowD = wave * 16 + 8 * hi;  // D-tile row base (+v)

  for (int t = 0; t < kT; ++t) {
    const int cur = t & 1;
    const float* hf_cur = hf + (size_t)cur * kB * kH;
    float*       hf_nxt = hf + (size_t)(cur ^ 1) * kB * kH;
    const u16*   hb_cur = hb + (size_t)cur * kB * kH;
    u16*         hb_nxt = hb + (size_t)(cur ^ 1) * kB * kH;

    // accumulators: [0,1]=r (i+h merged), [2,3]=z, [4,5]=n(x part), [6,7]=n(h part)
    v8f acc[8] = {};

    const u16* xrow = xb + ((size_t)mA * kT + t) * kIN;
    if (t + 1 < kT) __builtin_prefetch(xrow + kIN, 0, 1);  // next x_t -> global_prefetch_b8

    // ---- input projection: K = 256 of bf16 x ----
#pragma unroll
    for (int kt = 0; kt < kIN / 32; ++kt) {
      const int kb = kt * 32;
      BF16x16 a;
      a.q[0] = *(const uint4*)(xrow + kb + 16 * hi);
      a.q[1] = *(const uint4*)(xrow + kb + 16 * hi + 8);
#pragma unroll
      for (int n = 0; n < 6; ++n) {
        const u16* wp = wih_l + (n * 16 + lm) * kIN + kb + 16 * hi;
        BF16x16 b;
        b.q[0] = *(const uint4*)wp;
        b.q[1] = *(const uint4*)(wp + 8);
        acc[n] = __builtin_amdgcn_wmma_f32_16x16x32_bf16(
            false, a.v, false, b.v, (short)0, acc[n], false, false);
      }
    }

    // ---- recurrent projection: K = 512 of bf16 h ----
    const u16* hrow = hb_cur + (size_t)mA * kH;
#pragma unroll 8
    for (int kt = 0; kt < kH / 32; ++kt) {
      const int kb = kt * 32;
      BF16x16 a;
      a.q[0] = *(const uint4*)(hrow + kb + 16 * hi);
      a.q[1] = *(const uint4*)(hrow + kb + 16 * hi + 8);
#pragma unroll
      for (int n = 0; n < 6; ++n) {
        const u16* wp = whh_l + (n * 16 + lm) * kH + kb + 16 * hi;
        BF16x16 b;
        b.q[0] = *(const uint4*)wp;
        b.q[1] = *(const uint4*)(wp + 8);
        const int ai = (n < 4) ? n : n + 2;   // r,z merge; n-gate h-part separate
        acc[ai] = __builtin_amdgcn_wmma_f32_16x16x32_bf16(
            false, a.v, false, b.v, (short)0, acc[ai], false, false);
      }
    }

    // ---- gate math + h update (fp32) ----
#pragma unroll
    for (int tau = 0; tau < 2; ++tau) {
      const int col = c0 + 16 * tau + lm;
#pragma unroll
      for (int v = 0; v < 8; ++v) {
        const int mrow = mrowD + v;
        const float r  = sigmoidf(acc[tau][v]     + brg[tau]);
        const float z  = sigmoidf(acc[2 + tau][v] + bzg[tau]);
        const float nn = tanhf(acc[4 + tau][v] + bng[tau] +
                               r * (acc[6 + tau][v] + bnn[tau]));
        const float hold = hf_cur[(size_t)mrow * kH + col];
        const float hnew = (1.0f - z) * nn + z * hold;
        hf_nxt[(size_t)mrow * kH + col] = hnew;
        hb_nxt[(size_t)mrow * kH + col] = f2bf(hnew);
      }
    }

    // ---- device-scope release, grid barrier (monotonic counter), acquire ----
    __threadfence();
    __syncthreads();
    if (tid == 0) {
      __hip_atomic_fetch_add(syncp, 1u, __ATOMIC_ACQ_REL, __HIP_MEMORY_SCOPE_AGENT);
      const unsigned target = (unsigned)(kG * (t + 1));
      while (__hip_atomic_load(syncp, __ATOMIC_ACQUIRE, __HIP_MEMORY_SCOPE_AGENT) < target) {
        __builtin_amdgcn_s_sleep(8);
      }
    }
    __syncthreads();
    __threadfence();
  }
}

// ---------------------------------------------------------------------------
// Kernel 3: output head  out[b] = sigmoid(h_final[b,:] . W_out + b_out)
// ---------------------------------------------------------------------------
__global__ void gru_head(const float* __restrict__ hfin,   // [B][H] (buffer 0)
                         const float* __restrict__ W_out,  // [H]
                         const float* __restrict__ b_out,  // [1]
                         float* __restrict__ out) {
  const int b = blockIdx.x * blockDim.x + threadIdx.x;
  if (b >= kB) return;
  float s = 0.0f;
  const float* hr = hfin + (size_t)b * kH;
#pragma unroll 8
  for (int c = 0; c < kH; ++c) s += hr[c] * W_out[c];
  s += b_out[0];
  out[b] = sigmoidf(s);
}

// ---------------------------------------------------------------------------
// Launcher. ws layout (bytes):
//   [0,256)           grid-sync counter (memset to 0 each launch)
//   [256, +512K)      hf  : 2 x 128 x 512 fp32   (buf0 zero = h0)
//   [.. , +256K)      hb  : 2 x 128 x 512 bf16
//   [.. , +1536K)     whh_pk bf16
//   [.. , +768K)      wih_pk bf16
//   [.. , +64M)       xb  : (B,T,IN) bf16        total ~70 MB
// ---------------------------------------------------------------------------
extern "C" void kernel_launch(void* const* d_in, const int* in_sizes, int n_in,
                              void* d_out, int out_size, void* d_ws, size_t ws_size,
                              hipStream_t stream) {
  const float* x      = (const float*)d_in[0];
  const float* W_ih   = (const float*)d_in[1];
  const float* W_hh   = (const float*)d_in[2];
  const float* bias   = (const float*)d_in[3];
  const float* bias_n = (const float*)d_in[4];
  const float* W_out  = (const float*)d_in[5];
  const float* b_out  = (const float*)d_in[6];
  float* out = (float*)d_out;

  char* ws = (char*)d_ws;
  const size_t off_hf  = 256;
  const size_t off_hb  = off_hf + (size_t)2 * kB * kH * sizeof(float);
  const size_t off_whh = off_hb + (size_t)2 * kB * kH * sizeof(u16);
  const size_t off_wih = off_whh + (size_t)kG * kGC * kH * sizeof(u16);
  const size_t off_xb  = off_wih + (size_t)kG * kGC * kIN * sizeof(u16);

  unsigned* syncp = (unsigned*)ws;
  float* hf  = (float*)(ws + off_hf);
  u16*   hb  = (u16*)(ws + off_hb);
  u16*   whh = (u16*)(ws + off_whh);
  u16*   wih = (u16*)(ws + off_wih);
  u16*   xbw = (u16*)(ws + off_xb);

  // zero sync counter + both h buffers (h0 = 0); graph-capturable.
  hipMemsetAsync(ws, 0, off_whh, stream);

  // x -> bf16 (33,554,432 elems, 8 per thread -> 16384 blocks)
  const size_t nx = (size_t)kB * kT * kIN;
  gru_x_to_bf16<<<(int)(nx / 8 / 256), 256, 0, stream>>>(x, xbw);

  const int nconv = kG * kGC * kH;  // covers the smaller W_ih range too
  gru_pack_weights<<<(nconv + 255) / 256, 256, 0, stream>>>(W_ih, W_hh, wih, whh);

  const size_t lds_bytes = (size_t)(kGC * kH + kGC * kIN) * sizeof(u16);  // 144 KB
  hipFuncSetAttribute((const void*)gru_scan,
                      hipFuncAttributeMaxDynamicSharedMemorySize, (int)lds_bytes);
  gru_scan<<<kG, 256, lds_bytes, stream>>>(xbw, bias, bias_n, wih, whh, hf, hb, syncp);

  gru_head<<<1, kB, 0, stream>>>(hf, W_out, b_out, out);
}
